// myGPT2Attention_2774548873389
// MI455X (gfx1250) — matrix operand order repaired and checked by reading, not compiled, not run on hardware
//
#include <hip/hip_runtime.h>

// ---------------------------------------------------------------------------
// CDNA5 (gfx1250) fused centred-attention block, bf16 WMMA throughout.
// B=2, S=2048, E=1024, H=16, D=64.  M = B*S = 4096.
// Pipeline:
//   (a) one-time fp32->bf16 conversion of hidden + transposed weights
//   (b) fused QKV GEMM (WMMA, async-to-LDS staging)
//   (c) flash attention with centred terms (WMMA scores / P@V / mask@V)
//   (d) output projection GEMM (WMMA, async-to-LDS staging)
// ---------------------------------------------------------------------------

typedef __attribute__((ext_vector_type(16))) __bf16 v16bf;
typedef __attribute__((ext_vector_type(2)))  __bf16 v2bf;
typedef __attribute__((ext_vector_type(8)))  float  v8f;
typedef int v4i __attribute__((vector_size(16)));   // matches async-LDS builtin pointee

union Frag { v16bf v; unsigned u[8]; };

__device__ __forceinline__ v8f vzero() { v8f z = {}; return z; }

// fp32 pair -> packed bf16 dword (hardware cvt when available)
__device__ __forceinline__ unsigned pack2(float lo, float hi) {
#if defined(__AMDGCN__) && __has_builtin(__builtin_amdgcn_cvt_pk_bf16_f32)
  v2bf p = __builtin_amdgcn_cvt_pk_bf16_f32(lo, hi);
  return __builtin_bit_cast(unsigned, p);
#else
  unsigned ul = __builtin_bit_cast(unsigned, lo);
  unsigned uh = __builtin_bit_cast(unsigned, hi);
  ul += 0x7FFFu + ((ul >> 16) & 1u);
  uh += 0x7FFFu + ((uh >> 16) & 1u);
  return (ul >> 16) | (uh & 0xFFFF0000u);
#endif
}
__device__ __forceinline__ unsigned short f2bf(float f) {
  return (unsigned short)(pack2(f, 0.f) & 0xFFFFu);
}
__device__ __forceinline__ float bf2f(unsigned short h) {
  unsigned u = ((unsigned)h) << 16;
  return __builtin_bit_cast(float, u);
}
__device__ __forceinline__ v8f wmma_bf16(const Frag& a, const Frag& b, v8f c) {
  return __builtin_amdgcn_wmma_f32_16x16x32_bf16(false, a.v, false, b.v,
                                                 (short)0, c, false, false);
}

// ---- CDNA5 async global->LDS staging (16B per lane), with safe fallback ----
#if defined(__AMDGCN__) && __has_builtin(__builtin_amdgcn_global_load_async_to_lds_b128)
#define HAVE_ASYNC_LDS 1
#else
#define HAVE_ASYNC_LDS 0
#endif

__device__ __forceinline__ void stage_b128(void* lds, const void* g) {
#if HAVE_ASYNC_LDS
  __builtin_amdgcn_global_load_async_to_lds_b128(
      (__attribute__((address_space(1))) v4i*)(g),
      (__attribute__((address_space(3))) v4i*)(lds), 0, 0);
#else
  *(uint4*)lds = *(const uint4*)g;
#endif
}
__device__ __forceinline__ void async_fence() {
#if HAVE_ASYNC_LDS
#if __has_builtin(__builtin_amdgcn_s_wait_asynccnt)
  __builtin_amdgcn_s_wait_asynccnt(0);
#else
  asm volatile("s_wait_asynccnt 0x0" ::: "memory");
#endif
#endif
}

// ---------------------------------------------------------------------------
// One-time conversion kernels
// ---------------------------------------------------------------------------
__global__ __launch_bounds__(256)
void cvt_bf16_kernel(const float* __restrict__ src,
                     unsigned short* __restrict__ dst, int n) {
  const int i = (blockIdx.x * 256 + threadIdx.x) * 4;
  if (i + 3 < n) {
    uint2 p;
    p.x = pack2(src[i], src[i + 1]);
    p.y = pack2(src[i + 2], src[i + 3]);
    *(uint2*)(dst + i) = p;
  }
}

// wT[n][k] = bf16(scale * w[k][n]);  w is [K][N] fp32, wT is [N][K] bf16
__global__ __launch_bounds__(256)
void transpose_cvt_kernel(const float* __restrict__ w,
                          unsigned short* __restrict__ wT,
                          int K, int N, const float* __restrict__ scale_ptr) {
  __shared__ float tile[32][33];
  const float s = scale_ptr ? scale_ptr[0] : 1.f;
  const int kb = blockIdx.y * 32, nb = blockIdx.x * 32;
  const int tx = threadIdx.x & 31, ty = threadIdx.x >> 5;   // 32 x 8
#pragma unroll
  for (int i = ty; i < 32; i += 8)
    tile[i][tx] = w[(size_t)(kb + i) * N + nb + tx];
  __syncthreads();
#pragma unroll
  for (int i = ty; i < 32; i += 8)
    wT[(size_t)(nb + i) * K + kb + tx] = f2bf(s * tile[tx][i]);
}

// ---------------------------------------------------------------------------
// Kernel 1: fused QKV projection (all-bf16 operands).
//   cols 0..1023    -> Q = h @ Wq + bq   (stored * 1/8, exact in bf16)
//   cols 1024..2047 -> K = h @ Wk + bk
//   cols 2048..3071 -> V = g_vr*(h @ Wv) + g_vs*h   (also stored transposed)
// Block 256 threads = 8 waves; tile 128x128, K-tile 32; async LDS staging.
// ---------------------------------------------------------------------------
__global__ __launch_bounds__(256)
void qkv_gemm_kernel(const unsigned short* __restrict__ hbf,   // [4096][1024] bf16
                     const float* __restrict__ hidden,         // [4096][1024] f32
                     const unsigned short* __restrict__ wqkT,  // [2048][1024] bf16
                     const unsigned short* __restrict__ wvT,   // [1024][1024] bf16 (pre-scaled)
                     const float* __restrict__ qk_b,
                     const float* __restrict__ v_sg,
                     unsigned short* __restrict__ qws,
                     unsigned short* __restrict__ kws,
                     unsigned short* __restrict__ vws,
                     unsigned short* __restrict__ vtws)        // [32][64][2048]
{
  __shared__ unsigned As[128][16];   // [row][k-dword]
  __shared__ unsigned Bs[128][16];   // [col][k-dword]  (W^T rows)

  const int tid  = threadIdx.x;
  const int wid  = tid >> 5;
  const int lane = tid & 31;
  const int ln16 = lane & 15;
  const int hh   = lane >> 4;

  const int m0 = blockIdx.y * 128;
  const int n0 = blockIdx.x * 128;
  const int region = (n0 < 1024) ? 0 : (n0 < 2048 ? 1 : 2);   // Q / K / V

  const unsigned short* wT   = (region != 2) ? wqkT : wvT;
  const int             wrow = (region != 2) ? n0 : (n0 - 2048);

  const int wm = (wid >> 2) * 64;
  const int wn = (wid & 3) * 32;

  v8f acc[4][2];
#pragma unroll
  for (int i = 0; i < 4; ++i)
#pragma unroll
    for (int j = 0; j < 2; ++j) acc[i][j] = vzero();

  const int lr   = tid >> 1;          // 0..127
  const int half = (tid & 1) * 16;    // 0 / 16 (bf16 elems)

  for (int k0 = 0; k0 < 1024; k0 += 32) {
    const unsigned short* ga = hbf + (size_t)(m0 + lr) * 1024 + k0 + half;
    const unsigned short* gb = wT + (size_t)(wrow + lr) * 1024 + k0 + half;
    stage_b128(&As[lr][half / 2], ga);
    stage_b128(&As[lr][half / 2 + 4], ga + 8);
    stage_b128(&Bs[lr][half / 2], gb);
    stage_b128(&Bs[lr][half / 2 + 4], gb + 8);
    async_fence();
    __syncthreads();

    Frag bf[2];
#pragma unroll
    for (int ni = 0; ni < 2; ++ni) {
      const unsigned* brow = &Bs[wn + ni * 16 + ln16][0];
#pragma unroll
      for (int j = 0; j < 8; ++j) bf[ni].u[j] = brow[(j >> 2) * 8 + hh * 4 + (j & 3)];
    }
#pragma unroll
    for (int mi = 0; mi < 4; ++mi) {
      Frag a;
      const unsigned* arow = &As[wm + mi * 16 + ln16][0];
#pragma unroll
      for (int j = 0; j < 8; ++j) {
        const int grp = j >> 2, sub = j & 3;
        a.u[j] = arow[grp * 8 + hh * 4 + sub];
      }
#pragma unroll
      for (int ni = 0; ni < 2; ++ni)
        acc[mi][ni] = wmma_bf16(a, bf[ni], acc[mi][ni]);
    }
    __syncthreads();
  }

  const float vsg = (region == 2) ? v_sg[0] : 0.f;
#pragma unroll
  for (int mi = 0; mi < 4; ++mi)
#pragma unroll
    for (int ni = 0; ni < 2; ++ni) {
      const int gn = n0 + wn + ni * 16 + ln16;
      const float bias = (region == 2) ? 0.f : qk_b[gn];
#pragma unroll
      for (int r = 0; r < 8; ++r) {
        const int gm = m0 + wm + mi * 16 + r + hh * 8;
        const int b = gm >> 11, spos = gm & 2047;
        float val = acc[mi][ni][r] + bias;
        if (region == 0) {
          const int hd = gn >> 6, d = gn & 63;
          qws[(((size_t)(b * 16 + hd)) * 2048 + spos) * 64 + d] = f2bf(val * 0.125f);
        } else if (region == 1) {
          const int nn = gn - 1024, hd = nn >> 6, d = nn & 63;
          kws[(((size_t)(b * 16 + hd)) * 2048 + spos) * 64 + d] = f2bf(val);
        } else {
          const int nn = gn - 2048;
          val += vsg * hidden[(size_t)gm * 1024 + nn];
          const int hd = nn >> 6, d = nn & 63;
          const unsigned short bv = f2bf(val);
          vws[(((size_t)(b * 16 + hd)) * 2048 + spos) * 64 + d] = bv;
          vtws[(((size_t)(b * 16 + hd)) * 64 + d) * 2048 + spos] = bv;
        }
      }
    }
}

// ---------------------------------------------------------------------------
// Kernel 2: flash attention with centred-attention terms.
//   ctx[q] = g_r * softmax(QK^T masked) @ V + g_s*v[q] - g_c/(q+1)*sum_{k<=q} v[k]
// One wave = 16 queries; block = 8 waves; keys in chunks of 32.
// ---------------------------------------------------------------------------
__global__ __launch_bounds__(256)
void attn_kernel(const unsigned short* __restrict__ qws,   // [32][2048][64]
                 const unsigned short* __restrict__ kws,
                 const unsigned short* __restrict__ vws,
                 const unsigned short* __restrict__ vtws,  // [32][64][2048]
                 const float* __restrict__ grv,
                 const float* __restrict__ gsv,
                 const float* __restrict__ gcv,
                 unsigned short* __restrict__ cws)         // [2][2048][1024]
{
  __shared__ unsigned short Pst[8][16][34];

  const int tid  = threadIdx.x;
  const int wid  = tid >> 5;
  const int lane = tid & 31;
  const int ln16 = lane & 15;
  const int hh   = lane >> 4;

  const int bh   = blockIdx.y;
  const int head = bh & 15;
  const int qb   = blockIdx.x * 128 + wid * 16;
  const size_t rowbase = (size_t)bh * 2048;

  Frag qa[2];
  {
    const unsigned* qrow = (const unsigned*)(qws + (rowbase + qb + ln16) * 64);
#pragma unroll
    for (int j = 0; j < 8; ++j) {
      const int grp = j >> 2, sub = j & 3;
      qa[0].u[j] = qrow[0  + grp * 8 + hh * 4 + sub];
      qa[1].u[j] = qrow[16 + grp * 8 + hh * 4 + sub];
    }
  }

  float mrow[8], lrow[8];
  v8f cacc[4], c2acc[4];
#pragma unroll
  for (int r = 0; r < 8; ++r) { mrow[r] = -3.0e38f; lrow[r] = 0.f; }
#pragma unroll
  for (int t = 0; t < 4; ++t) { cacc[t] = vzero(); c2acc[t] = vzero(); }

  for (int kb = 0; kb < qb + 16; kb += 32) {
    // ---- scores 16x32 ----
    v8f s[2];
#pragma unroll
    for (int t = 0; t < 2; ++t) {
      s[t] = vzero();
      Frag kf0, kf1;
      const unsigned* krow = (const unsigned*)(kws + (rowbase + kb + t * 16 + ln16) * 64);
#pragma unroll
      for (int j = 0; j < 8; ++j) { kf0.u[j] = krow[(j >> 2) * 8 + hh * 4 + (j & 3)]; kf1.u[j] = krow[16 + (j >> 2) * 8 + hh * 4 + (j & 3)]; }
      s[t] = wmma_bf16(qa[0], kf0, s[t]);
      s[t] = wmma_bf16(qa[1], kf1, s[t]);
    }

    // ---- causal mask + online softmax ----
#pragma unroll
    for (int r = 0; r < 8; ++r) {
      const int q = qb + r + hh * 8;
#pragma unroll
      for (int t = 0; t < 2; ++t) {
        const int key = kb + t * 16 + ln16;
        if (key > q) s[t][r] = -3.0e38f;
      }
      float mx = fmaxf(s[0][r], s[1][r]);
      mx = fmaxf(mx, __shfl_xor(mx, 1, 32));
      mx = fmaxf(mx, __shfl_xor(mx, 2, 32));
      mx = fmaxf(mx, __shfl_xor(mx, 4, 32));
      mx = fmaxf(mx, __shfl_xor(mx, 8, 32));
      const float mnew = fmaxf(mrow[r], mx);
      const float sc = __expf(mrow[r] - mnew);
      mrow[r] = mnew;
      const float p0 = __expf(s[0][r] - mnew);
      const float p1 = __expf(s[1][r] - mnew);
      s[0][r] = p0; s[1][r] = p1;
      float rs = p0 + p1;
      rs += __shfl_xor(rs, 1, 32);
      rs += __shfl_xor(rs, 2, 32);
      rs += __shfl_xor(rs, 4, 32);
      rs += __shfl_xor(rs, 8, 32);
      lrow[r] = lrow[r] * sc + rs;
#pragma unroll
      for (int t = 0; t < 4; ++t) cacc[t][r] *= sc;
    }

    // ---- P: C-layout -> A-layout via per-wave LDS; causal ones-mask frag ----
#pragma unroll
    for (int t = 0; t < 2; ++t)
#pragma unroll
      for (int r = 0; r < 8; ++r)
        Pst[wid][r + hh * 8][t * 16 + ln16] = f2bf(s[t][r]);

    Frag pa, ma;
    {
      const unsigned short* prow = &Pst[wid][ln16][0];
      const int q = qb + ln16;
#pragma unroll
      for (int j = 0; j < 8; ++j) {
        const int grp = j >> 2, sub = j & 3;
        const int kk = hh * 16 + j * 2;
        pa.u[j] = *(const unsigned*)(prow + kk);
        const int key0 = kb + kk;
        const unsigned lo = (key0     <= q) ? 0x3F80u : 0u;
        const unsigned hi = (key0 + 1 <= q) ? 0x3F80u : 0u;
        ma.u[j] = lo | (hi << 16);
      }
    }

    // ---- V B-fragments from V^T (contiguous b128 loads) ----
#pragma unroll
    for (int dt = 0; dt < 4; ++dt) {
      Frag vb;
      const unsigned* vt = (const unsigned*)(vtws
          + ((size_t)bh * 64 + dt * 16 + ln16) * 2048 + kb + hh * 16);
#pragma unroll
      for (int j = 0; j < 8; ++j) vb.u[j] = vt[j];
      cacc[dt]  = wmma_bf16(pa, vb, cacc[dt]);
      c2acc[dt] = wmma_bf16(ma, vb, c2acc[dt]);
    }
  }

  // ---- epilogue ----
  const float gr = grv[head], gs = gsv[head], gc = gcv[head];
  const int b = bh >> 4;
#pragma unroll
  for (int dt = 0; dt < 4; ++dt) {
    const int d = dt * 16 + ln16;
#pragma unroll
    for (int r = 0; r < 8; ++r) {
      const int q = qb + r + hh * 8;
      const float vq = bf2f(vws[(rowbase + q) * 64 + d]);
      const float val = gr * (cacc[dt][r] / lrow[r])
                      + gs * vq
                      - gc * (c2acc[dt][r] / (float)(q + 1));
      cws[((size_t)b * 2048 + q) * 1024 + head * 64 + d] = f2bf(val);
    }
  }
}

// ---------------------------------------------------------------------------
// Kernel 3: output projection.  out = ctx @ (g_pr * Wp) + g_ps * ctx  (fp32)
// ---------------------------------------------------------------------------
__global__ __launch_bounds__(256)
void proj_gemm_kernel(const unsigned short* __restrict__ ctxb,  // [4096][1024]
                      const unsigned short* __restrict__ wpT,   // [1024][1024] pre-scaled
                      const float* __restrict__ p_sg,
                      float* __restrict__ out)
{
  __shared__ unsigned As[128][16];
  __shared__ unsigned Bs[128][16];

  const int tid  = threadIdx.x;
  const int wid  = tid >> 5;
  const int lane = tid & 31;
  const int ln16 = lane & 15;
  const int hh   = lane >> 4;

  const int m0 = blockIdx.y * 128;
  const int n0 = blockIdx.x * 128;

  const int wm = (wid >> 2) * 64;
  const int wn = (wid & 3) * 32;

  v8f acc[4][2];
#pragma unroll
  for (int i = 0; i < 4; ++i)
#pragma unroll
    for (int j = 0; j < 2; ++j) acc[i][j] = vzero();

  const int lr   = tid >> 1;
  const int half = (tid & 1) * 16;

  for (int k0 = 0; k0 < 1024; k0 += 32) {
    const unsigned short* ga = ctxb + (size_t)(m0 + lr) * 1024 + k0 + half;
    const unsigned short* gb = wpT + (size_t)(n0 + lr) * 1024 + k0 + half;
    stage_b128(&As[lr][half / 2], ga);
    stage_b128(&As[lr][half / 2 + 4], ga + 8);
    stage_b128(&Bs[lr][half / 2], gb);
    stage_b128(&Bs[lr][half / 2 + 4], gb + 8);
    async_fence();
    __syncthreads();

    Frag bf[2];
#pragma unroll
    for (int ni = 0; ni < 2; ++ni) {
      const unsigned* brow = &Bs[wn + ni * 16 + ln16][0];
#pragma unroll
      for (int j = 0; j < 8; ++j) bf[ni].u[j] = brow[(j >> 2) * 8 + hh * 4 + (j & 3)];
    }
#pragma unroll
    for (int mi = 0; mi < 4; ++mi) {
      Frag a;
      const unsigned* arow = &As[wm + mi * 16 + ln16][0];
#pragma unroll
      for (int j = 0; j < 8; ++j) {
        const int grp = j >> 2, sub = j & 3;
        a.u[j] = arow[grp * 8 + hh * 4 + sub];
      }
#pragma unroll
      for (int ni = 0; ni < 2; ++ni)
        acc[mi][ni] = wmma_bf16(a, bf[ni], acc[mi][ni]);
    }
    __syncthreads();
  }

  const float psg = p_sg[0];
#pragma unroll
  for (int mi = 0; mi < 4; ++mi)
#pragma unroll
    for (int ni = 0; ni < 2; ++ni) {
      const int gn = n0 + wn + ni * 16 + ln16;
#pragma unroll
      for (int r = 0; r < 8; ++r) {
        const int gm = m0 + wm + mi * 16 + r + hh * 8;
        const float skip = psg * bf2f(ctxb[(size_t)gm * 1024 + gn]);
        out[(size_t)gm * 1024 + gn] = acc[mi][ni][r] + skip;
      }
    }
}

// ---------------------------------------------------------------------------
// Host-side launcher
// ---------------------------------------------------------------------------
extern "C" void kernel_launch(void* const* d_in, const int* in_sizes, int n_in,
                              void* d_out, int out_size, void* d_ws, size_t ws_size,
                              hipStream_t stream) {
  (void)in_sizes; (void)n_in; (void)out_size; (void)ws_size;

  const float* hidden = (const float*)d_in[0];
  const float* qk_w   = (const float*)d_in[1];
  const float* qk_b   = (const float*)d_in[2];
  const float* v_w    = (const float*)d_in[3];
  const float* v_rg   = (const float*)d_in[4];
  const float* v_sg   = (const float*)d_in[5];
  const float* p_w    = (const float*)d_in[6];
  const float* p_rg   = (const float*)d_in[7];
  const float* p_sg   = (const float*)d_in[8];
  const float* a_rg   = (const float*)d_in[9];
  const float* a_sg   = (const float*)d_in[10];
  const float* a_cg   = (const float*)d_in[11];
  float* out = (float*)d_out;

  const size_t SZ = (size_t)4096 * 1024;          // 4M bf16 elements
  unsigned short* qws  = (unsigned short*)d_ws;
  unsigned short* kws  = qws  + SZ;
  unsigned short* vws  = kws  + SZ;
  unsigned short* vtws = vws  + SZ;
  unsigned short* cws  = vtws + SZ;
  unsigned short* hbf  = cws  + SZ;
  unsigned short* wqkT = hbf  + SZ;                // 2M
  unsigned short* wvT  = wqkT + (size_t)2048 * 1024;
  unsigned short* wpT  = wvT  + (size_t)1024 * 1024;

  dim3 blk(256, 1, 1);

  // one-time conversions
  cvt_bf16_kernel<<<4096, blk, 0, stream>>>(hidden, hbf, 4096 * 1024);
  transpose_cvt_kernel<<<dim3(64, 32), blk, 0, stream>>>(qk_w, wqkT, 1024, 2048, nullptr);
  transpose_cvt_kernel<<<dim3(32, 32), blk, 0, stream>>>(v_w,  wvT,  1024, 1024, v_rg);
  transpose_cvt_kernel<<<dim3(32, 32), blk, 0, stream>>>(p_w,  wpT,  1024, 1024, p_rg);

  // main pipeline
  qkv_gemm_kernel<<<dim3(24, 32), blk, 0, stream>>>(hbf, hidden, wqkT, wvT,
                                                    qk_b, v_sg, qws, kws, vws, vtws);
  attn_kernel<<<dim3(16, 32), blk, 0, stream>>>(qws, kws, vws, vtws,
                                                a_rg, a_sg, a_cg, cws);
  proj_gemm_kernel<<<dim3(8, 32), blk, 0, stream>>>(cws, wpT, p_sg, out);
}
